// WindowAttention3D_58677843198576
// MI455X (gfx1250) — matrix operand order repaired and checked by reading, not compiled, not run on hardware
//
#include <hip/hip_runtime.h>
#include <hip/hip_bf16.h>

typedef __attribute__((ext_vector_type(16))) _Float16 v16h;
typedef __attribute__((ext_vector_type(8)))  _Float16 v8h;
typedef __attribute__((ext_vector_type(8)))  float    v8f;

__device__ __forceinline__ v8f wmma_f16(v16h a, v16h b, v8f c) {
  return __builtin_amdgcn_wmma_f32_16x16x32_f16(false, a, false, b, (short)0, c, false, false);
}

__device__ __forceinline__ v16h cat8(v8h lo, v8h hi) {
  v16h r;
#pragma unroll
  for (int e = 0; e < 8; ++e) { r[e] = lo[e]; r[e + 8] = hi[e]; }
  return r;
}

__device__ __forceinline__ void wait_async0() {
#if __has_builtin(__builtin_amdgcn_s_wait_asynccnt)
  __builtin_amdgcn_s_wait_asynccnt(0);
#else
  asm volatile("s_wait_asynccnt 0x0" ::: "memory");
#endif
}

// ---------------------------------------------------------------------------
// Pre-convert weights to f16 (done once; removes per-tile cvt in GEMM loops)
// ---------------------------------------------------------------------------
__global__ void wcvt_kernel(const float* __restrict__ qkv_w,
                            const float* __restrict__ proj_w,
                            _Float16* __restrict__ qkv_wh,
                            _Float16* __restrict__ proj_wh) {
  int i = blockIdx.x * blockDim.x + threadIdx.x;
  if (i < 576 * 192) qkv_wh[i] = (_Float16)qkv_w[i];
  if (i < 192 * 192) proj_wh[i] = (_Float16)proj_w[i];
}

// ---------------------------------------------------------------------------
// CPB MLP: tbl[r, head] = relu(table[r] @ w1^T + b1) @ w2^T   (507 x 6)
// ---------------------------------------------------------------------------
__global__ void cpb_tbl_kernel(const float* __restrict__ table,
                               const float* __restrict__ w1,
                               const float* __restrict__ b1,
                               const float* __restrict__ w2,
                               float* __restrict__ tbl) {
  int r = blockIdx.x * blockDim.x + threadIdx.x;
  if (r >= 507) return;
  float c0 = table[r * 3 + 0], c1 = table[r * 3 + 1], c2 = table[r * 3 + 2];
  float acc[6] = {0.f, 0.f, 0.f, 0.f, 0.f, 0.f};
  for (int j = 0; j < 512; ++j) {
    float hh = b1[j] + c0 * w1[j * 3] + c1 * w1[j * 3 + 1] + c2 * w1[j * 3 + 2];
    hh = fmaxf(hh, 0.f);
#pragma unroll
    for (int hd = 0; hd < 6; ++hd) acc[hd] += hh * w2[hd * 512 + j];
  }
#pragma unroll
  for (int hd = 0; hd < 6; ++hd) tbl[r * 6 + hd] = acc[hd];
}

__global__ void cpb_bias_kernel(const float* __restrict__ tbl,
                                const int* __restrict__ rpi,
                                float* __restrict__ biasb) {
  int i = blockIdx.x * blockDim.x + threadIdx.x;
  if (i >= 6 * 98 * 98) return;
  int h = i / 9604;
  int nm = i % 9604;
  float v = tbl[rpi[nm] * 6 + h];
  biasb[i] = 16.f / (1.f + __expf(-v));
}

// ---------------------------------------------------------------------------
// QKV projection: qkvh[b,0..111,0..575] (f16) = x @ qkv_w^T + qkv_b
// x async-DMA'd to LDS (f32), converted once to f16 LDS; weights pre-f16.
// ---------------------------------------------------------------------------
__global__ __launch_bounds__(256) void qkv_kernel(const float* __restrict__ x,
                                                  const _Float16* __restrict__ qkv_wh,
                                                  const float* __restrict__ qkv_b,
                                                  _Float16* __restrict__ qkvh) {
  extern __shared__ char smem[];
  float* Sx = (float*)smem;                       // 98*192 f32 = 75264 B
  _Float16* xh = (_Float16*)(smem + 75264);       // 112*192 f16 = 43008 B
  const int b = blockIdx.x;
  const int tid = threadIdx.x;

  // async DMA: 98*192 floats = 4704 chunks of 16B, global -> LDS
  {
    unsigned ldsbase = (unsigned)(unsigned long long)(void*)Sx;
    unsigned long long gbase =
        (unsigned long long)(const void*)(x + (size_t)b * 18816);
    for (int c = tid; c < 4704; c += 256) {
      unsigned lo = ldsbase + 16u * (unsigned)c;
      unsigned long long ga = gbase + 16ull * (unsigned)c;
      asm volatile("global_load_async_to_lds_b128 %0, %1, off"
                   :: "v"(lo), "v"(ga) : "memory");
    }
    wait_async0();
  }
  __syncthreads();
  // single conversion pass f32 -> f16 (pad rows 98..111 zeroed)
  for (int c = tid; c < 2688; c += 256) {  // 8-half chunks, 21504 halves
    v8h v = {};
    if (c < 2352) {  // 18816/8: real data
      const float4* p = (const float4*)(Sx + c * 8);
      float4 f0 = p[0], f1 = p[1];
      v[0] = (_Float16)f0.x; v[1] = (_Float16)f0.y; v[2] = (_Float16)f0.z; v[3] = (_Float16)f0.w;
      v[4] = (_Float16)f1.x; v[5] = (_Float16)f1.y; v[6] = (_Float16)f1.z; v[7] = (_Float16)f1.w;
    }
    *(v8h*)(xh + c * 8) = v;
  }
  __syncthreads();

  const int lane = tid & 31, wave = tid >> 5;
  const int mrow = lane & 15, hl = lane >> 4;
  for (int t = wave; t < 7 * 36; t += 8) {
    int m0 = (t / 36) * 16, n0 = (t % 36) * 16;
    v8f acc = {};
    const v8h* arow = (const v8h*)(xh + (m0 + mrow) * 192);
    const v8h* brow = (const v8h*)(qkv_wh + (size_t)(n0 + mrow) * 192);
#pragma unroll
    for (int kc = 0; kc < 6; ++kc) {            // k0 = kc*32
      v16h a = cat8(arow[kc * 4 + hl * 2], arow[kc * 4 + hl * 2 + 1]);
      v16h bb = cat8(brow[kc * 4 + hl * 2], brow[kc * 4 + hl * 2 + 1]);
      acc = wmma_f16(a, bb, acc);
    }
#pragma unroll
    for (int r = 0; r < 8; ++r) {
      int row = m0 + r + (hl << 3);
      int col = n0 + mrow;
      qkvh[((size_t)b * 112 + row) * 576 + col] = (_Float16)(acc[r] + qkv_b[col]);
    }
  }
}

// ---------------------------------------------------------------------------
// Per-row inverse norms; q side folded with exp(min(logit_scale, log 100))
// ---------------------------------------------------------------------------
__global__ void norm_kernel(const _Float16* __restrict__ qkvh,
                            const float* __restrict__ logit_scale,
                            float* __restrict__ qs, float* __restrict__ ks) {
  int i = blockIdx.x * blockDim.x + threadIdx.x;
  if (i >= 2048 * 6 * 112) return;
  int n = i % 112;
  int h = (i / 112) % 6;
  int b = i / (6 * 112);
  float qv = 0.f, kv = 0.f;
  if (n < 98) {
    const v8h* q8 = (const v8h*)(qkvh + ((size_t)b * 112 + n) * 576 + h * 32);
    const v8h* k8 = q8 + 24;  // +192 halves
    float sq = 0.f, sk = 0.f;
#pragma unroll
    for (int c = 0; c < 4; ++c) {
      v8h qc = q8[c], kc = k8[c];
#pragma unroll
      for (int e = 0; e < 8; ++e) {
        float q = (float)qc[e]; sq += q * q;
        float k = (float)kc[e]; sk += k * k;
      }
    }
    float sc = __expf(fminf(logit_scale[h], 4.6051702f));  // log(100)
    qv = sc / fmaxf(sqrtf(sq), 1e-12f);
    kv = 1.f / fmaxf(sqrtf(sk), 1e-12f);
  }
  qs[i] = qv;
  ks[i] = kv;
}

// ---------------------------------------------------------------------------
// Fused attention + output projection, one block per window.
// LDS: S (112x112 f32 scores), P (112x128 f16 probs, pad cols zero),
//      O (112x192 f16 head outputs), Vt (32x112 f16, per-head V^T)
// ---------------------------------------------------------------------------
__global__ __launch_bounds__(256) void attn_kernel(const _Float16* __restrict__ qkvh,
                                                   const float* __restrict__ qs,
                                                   const float* __restrict__ ks,
                                                   const float* __restrict__ biasb,
                                                   const float* __restrict__ mask,
                                                   const _Float16* __restrict__ proj_wh,
                                                   const float* __restrict__ proj_b,
                                                   float* __restrict__ out) {
  extern __shared__ char smem[];
  float* S = (float*)smem;                                    // 50176 B
  _Float16* P = (_Float16*)(smem + 50176);                    // 28672 B
  _Float16* O = (_Float16*)(smem + 50176 + 28672);            // 43008 B
  _Float16* Vt = (_Float16*)(smem + 50176 + 28672 + 43008);   //  7168 B
  const int b = blockIdx.x;
  const int wimg = b % 64;
  const int tid = threadIdx.x;
  const int lane = tid & 31, wave = tid >> 5;
  const int mrow = lane & 15, hl = lane >> 4;
  const size_t base = (size_t)b * 112 * 576;

  // zero pad columns 112..127 of P once (stay zero across heads)
  for (int i = tid; i < 112 * 16; i += 256) {
    int row = i >> 4;
    P[row * 128 + 112 + (i & 15)] = (_Float16)0.f;
  }

  for (int h = 0; h < 6; ++h) {
    // per-head V^T into LDS: Vt[d*112 + token], d = head-local dim 0..31
    for (int i = tid; i < 112 * 4; i += 256) {
      int m = i >> 2, dc = i & 3;
      v8h v = *(const v8h*)(qkvh + base + (size_t)m * 576 + 384 + h * 32 + dc * 8);
#pragma unroll
      for (int e = 0; e < 8; ++e) Vt[(dc * 8 + e) * 112 + m] = v[e];
    }
    const float* qsr = qs + ((size_t)b * 6 + h) * 112;
    const float* ksr = ks + ((size_t)b * 6 + h) * 112;
    // ---- scores = (q @ k^T) * qs x ks + bias + mask --------------------
    for (int t = wave; t < 49; t += 8) {
      int m0 = (t / 7) * 16, n0 = (t % 7) * 16;
      const v8h* pq = (const v8h*)(qkvh + base + (size_t)(m0 + mrow) * 576 + h * 32);
      v16h a = cat8(pq[2 * hl], pq[2 * hl + 1]);
      const v8h* pk = (const v8h*)(qkvh + base + (size_t)(n0 + mrow) * 576 + 192 + h * 32);
      v16h bb = cat8(pk[2 * hl], pk[2 * hl + 1]);
      v8f acc = {};
      acc = wmma_f16(a, bb, acc);
#pragma unroll
      for (int r = 0; r < 8; ++r) {
        int row = m0 + r + (hl << 3);
        int col = n0 + mrow;
        float s = acc[r] * qsr[row] * ksr[col];
        if (row < 98 && col < 98)
          s += biasb[((size_t)h * 98 + row) * 98 + col] +
               mask[((size_t)wimg * 98 + row) * 98 + col];
        if (col >= 98) s = -1e30f;
        S[row * 112 + col] = s;
      }
    }
    __syncthreads();
    // ---- softmax rows (f32 in S) -> f16 normalized probs in P ----------
    for (int row = wave; row < 112; row += 8) {
      float v0[4];
      float mx = -1e30f;
#pragma unroll
      for (int j = 0; j < 4; ++j) {
        int c = lane + 32 * j;
        v0[j] = (c < 112) ? S[row * 112 + c] : -1e30f;
        mx = fmaxf(mx, v0[j]);
      }
      for (int off = 16; off > 0; off >>= 1) mx = fmaxf(mx, __shfl_xor(mx, off, 32));
      float sum = 0.f;
#pragma unroll
      for (int j = 0; j < 4; ++j) {
        v0[j] = __expf(v0[j] - mx);
        if (lane + 32 * j >= 112) v0[j] = 0.f;
        sum += v0[j];
      }
      for (int off = 16; off > 0; off >>= 1) sum += __shfl_xor(sum, off, 32);
      float inv = 1.f / sum;
#pragma unroll
      for (int j = 0; j < 4; ++j) {
        int c = lane + 32 * j;
        if (c < 112) P[row * 128 + c] = (_Float16)(v0[j] * inv);
      }
    }
    __syncthreads();
    // ---- O_h = P @ V : pure b128 fragment loads ------------------------
    for (int t = wave; t < 14; t += 8) {
      int m0 = (t / 2) * 16, n0 = (t % 2) * 16;
      v8f acc = {};
      const v8h* pp = (const v8h*)(P + (m0 + mrow) * 128);
      const _Float16* vrow = Vt + (n0 + mrow) * 112;
      const v8h zh = {};
#pragma unroll
      for (int kc = 0; kc < 4; ++kc) {  // k0 = kc*32
        v16h a = cat8(pp[kc * 4 + hl * 2], pp[kc * 4 + hl * 2 + 1]);
        int tk = kc * 32 + hl * 16;     // 16 consecutive tokens of V^T
        const v8h* pv = (const v8h*)(vrow + tk);
        v16h bb = (tk < 112) ? cat8(pv[0], pv[1]) : cat8(zh, zh);
        acc = wmma_f16(a, bb, acc);
      }
#pragma unroll
      for (int r = 0; r < 8; ++r) {
        int row = m0 + r + (hl << 3);
        O[row * 192 + h * 32 + n0 + mrow] = (_Float16)acc[r];
      }
    }
    __syncthreads();  // protect S/P/Vt before next head overwrites them
  }
  // ---- final projection: out = O @ proj_w^T + proj_b --------------------
  for (int t = wave; t < 84; t += 8) {
    int m0 = (t / 12) * 16, n0 = (t % 12) * 16;
    v8f acc = {};
    const v8h* po = (const v8h*)(O + (m0 + mrow) * 192);
    const v8h* pw = (const v8h*)(proj_wh + (size_t)(n0 + mrow) * 192);
#pragma unroll
    for (int kc = 0; kc < 6; ++kc) {  // k0 = kc*32
      v16h a = cat8(po[kc * 4 + hl * 2], po[kc * 4 + hl * 2 + 1]);
      v16h bb = cat8(pw[kc * 4 + hl * 2], pw[kc * 4 + hl * 2 + 1]);
      acc = wmma_f16(a, bb, acc);
    }
#pragma unroll
    for (int r = 0; r < 8; ++r) {
      int row = m0 + r + (hl << 3);
      int col = n0 + mrow;
      if (row < 98)
        out[((size_t)b * 98 + row) * 192 + col] = acc[r] + proj_b[col];
    }
  }
}

// ---------------------------------------------------------------------------
extern "C" void kernel_launch(void* const* d_in, const int* in_sizes, int n_in,
                              void* d_out, int out_size, void* d_ws, size_t ws_size,
                              hipStream_t stream) {
  const float* x           = (const float*)d_in[0];
  const float* mask        = (const float*)d_in[1];
  const float* qkv_w       = (const float*)d_in[2];
  const float* qkv_b       = (const float*)d_in[3];
  const float* logit_scale = (const float*)d_in[4];
  const float* cpb_w1      = (const float*)d_in[5];
  const float* cpb_b1      = (const float*)d_in[6];
  const float* cpb_w2      = (const float*)d_in[7];
  const float* proj_w      = (const float*)d_in[8];
  const float* proj_b      = (const float*)d_in[9];
  const float* table       = (const float*)d_in[10];
  const int*   rpi         = (const int*)d_in[11];
  float* out = (float*)d_out;

  char* w = (char*)d_ws;
  size_t off = 0;
  _Float16* qkvh   = (_Float16*)(w + off); off += (size_t)2048 * 112 * 576 * 2; // 264 MB
  float* qs        = (float*)(w + off);    off += (size_t)2048 * 6 * 112 * 4;
  float* ks        = (float*)(w + off);    off += (size_t)2048 * 6 * 112 * 4;
  float* biasb     = (float*)(w + off);    off += (size_t)6 * 98 * 98 * 4;
  float* tbl       = (float*)(w + off);    off += (size_t)507 * 6 * 4;
  _Float16* qkv_wh = (_Float16*)(w + off); off += (size_t)576 * 192 * 2;
  _Float16* proj_wh= (_Float16*)(w + off); off += (size_t)192 * 192 * 2;
  (void)ws_size; (void)in_sizes; (void)n_in; (void)out_size;

  wcvt_kernel<<<(576 * 192 + 255) / 256, 256, 0, stream>>>(qkv_w, proj_w, qkv_wh, proj_wh);
  cpb_tbl_kernel<<<2, 256, 0, stream>>>(table, cpb_w1, cpb_b1, cpb_w2, tbl);
  cpb_bias_kernel<<<(6 * 98 * 98 + 255) / 256, 256, 0, stream>>>(tbl, rpi, biasb);
  qkv_kernel<<<2048, 256, 75264 + 43008, stream>>>(x, qkv_wh, qkv_b, qkvh);
  norm_kernel<<<(2048 * 6 * 112 + 255) / 256, 256, 0, stream>>>(qkvh, logit_scale, qs, ks);
  attn_kernel<<<2048, 256, 50176 + 28672 + 43008 + 7168, stream>>>(
      qkvh, qs, ks, biasb, mask, proj_wh, proj_b, out);
}
